// VolumeRenderer_53472342835699
// MI455X (gfx1250) — matrix operand order repaired and checked by reading, compiled, not run on hardware
//
#include <hip/hip_runtime.h>
#include <hip/hip_bf16.h>
#include <math.h>

typedef __attribute__((ext_vector_type(16))) _Float16 v16h;
typedef __attribute__((ext_vector_type(8)))  float    v8f;

#define NEARV 0.5f
#define FARV  2.5f
#define NS    64
#define NRAYS 4096
#define NPTS  (NRAYS * NS)

// Workspace layout (bytes, all offsets 256B aligned)
#define WT1_OFF 0u         // 256*32  f16 = 16384
#define WT2_OFF 16384u     // 256*256 f16 = 131072
#define WT3_OFF 147456u    // 256*256 f16 = 131072
#define WT4_OFF 278528u    // 128*288 f16 = 73728
#define DEN_OFF 352256u    // 262144 f32  = 1048576
#define RGB_OFF 1400832u   // 262144*3 f32 = 3145728
// total ~4.34 MB

union Frag { v16h v; uint4 q[2]; };

// A fragment (16x32 f16, M x K). ISA layout: lane<16 -> row=lane, halves 0-7 = K kb+0..7,
// halves 8-15 = K kb+16..23; lane>=16 -> row=lane-16, K kb+8..15 and kb+24..31.
__device__ __forceinline__ v16h load_a_frag(const _Float16* base, int row, int strideH,
                                            int kb, int hi) {
  Frag f;
  const _Float16* p = base + row * strideH + kb + hi * 8;
  f.q[0] = *(const uint4*)(p);
  f.q[1] = *(const uint4*)(p + 16);
  return f.v;
}

// B fragment (32x16 f16, K x N) from transposed weights Wt[out][in]:
// lane<16 -> col=lane holds K kb+0..7, kb+16..23; lane>=16 -> col=lane-16 holds K kb+8..15, kb+24..31.
__device__ __forceinline__ v16h load_b_frag(const _Float16* w, int col, int strideH,
                                            int kb, int hi) {
  Frag f;
  const _Float16* p = w + col * strideH + kb + hi * 8;
  f.q[0] = *(const uint4*)(p);
  f.q[1] = *(const uint4*)(p + 16);
  return f.v;
}

// One MLP layer over a 32-point LDS tile: Out[32 x 16*NT*8] = relu(A @ Wt^T + bias).
// 2x NT register blocking: each wave owns BOTH 16-row M-tiles and NT consecutive
// 16-col N-tiles (base wave*NT), so every weight (B) fragment is fetched exactly once
// per block, and each K-step issues 2*NT independent WMMAs (accumulator-disjoint) that
// cover the WMMA->WMMA hazard window. K = 32*KT. If Atail != nullptr, the final
// K-tile's A comes from Atail (zero-padded viewdirs concat tail for layer 4).
template <int NT, int KT>
__device__ __forceinline__ void gemm_layer(
    const _Float16* __restrict__ A, int aStride,
    const _Float16* __restrict__ Atail, int tailStride,
    const _Float16* __restrict__ Wt, int wStride,
    const float* __restrict__ bias,
    _Float16* __restrict__ Out, int outStride,
    int wave, int lane)
{
  const int lane16 = lane & 15;
  const int hi     = lane >> 4;
  const int ntb    = wave * NT;

  v8f c[2][NT];
#pragma unroll
  for (int j = 0; j < NT; ++j) {
    const float bv = bias[(ntb + j) * 16 + lane16];
#pragma unroll
    for (int r = 0; r < 8; ++r) { c[0][j][r] = bv; c[1][j][r] = bv; }
  }

#pragma unroll 2
  for (int kt = 0; kt < KT; ++kt) {
    v16h a0, a1;
    if (Atail != nullptr && kt == KT - 1) {
      a0 = load_a_frag(Atail, lane16,      tailStride, 0, hi);
      a1 = load_a_frag(Atail, 16 + lane16, tailStride, 0, hi);
    } else {
      a0 = load_a_frag(A, lane16,      aStride, kt * 32, hi);
      a1 = load_a_frag(A, 16 + lane16, aStride, kt * 32, hi);
    }
#pragma unroll
    for (int j = 0; j < NT; ++j) {
      v16h b = load_b_frag(Wt, (ntb + j) * 16 + lane16, wStride, kt * 32, hi);
      c[0][j] = __builtin_amdgcn_wmma_f32_16x16x32_f16(
          false, a0, false, b, (short)0, c[0][j], false, false);
      c[1][j] = __builtin_amdgcn_wmma_f32_16x16x32_f16(
          false, a1, false, b, (short)0, c[1][j], false, false);
    }
  }

  // fused ReLU + f16 store; D layout: VGPR r, lanes 0-15 -> M=r, lanes 16-31 -> M=r+8
#pragma unroll
  for (int mt = 0; mt < 2; ++mt) {
#pragma unroll
    for (int j = 0; j < NT; ++j) {
#pragma unroll
      for (int r = 0; r < 8; ++r) {
        float vv = c[mt][j][r];
        vv = vv > 0.f ? vv : 0.f;
        Out[(mt * 16 + hi * 8 + r) * outStride + (ntb + j) * 16 + lane16] = (_Float16)vv;
      }
    }
  }
}

// ---------------- kernel 1: weight transpose + f32 -> f16 ----------------
__global__ void prep_weights_kernel(const float* __restrict__ W1, const float* __restrict__ W2,
                                    const float* __restrict__ W3, const float* __restrict__ W4,
                                    _Float16* __restrict__ Wt1, _Float16* __restrict__ Wt2,
                                    _Float16* __restrict__ Wt3, _Float16* __restrict__ Wt4) {
  const int total = 8192 + 65536 + 65536 + 36864;
  for (int idx = blockIdx.x * blockDim.x + threadIdx.x; idx < total;
       idx += gridDim.x * blockDim.x) {
    if (idx < 8192) {
      int e = idx, n = e >> 5, k = e & 31;
      Wt1[n * 32 + k] = (_Float16)((k < 3) ? W1[k * 256 + n] : 0.f);
    } else if (idx < 8192 + 65536) {
      int e = idx - 8192, n = e >> 8, k = e & 255;
      Wt2[n * 256 + k] = (_Float16)W2[k * 256 + n];
    } else if (idx < 8192 + 65536 + 65536) {
      int e = idx - 8192 - 65536, n = e >> 8, k = e & 255;
      Wt3[n * 256 + k] = (_Float16)W3[k * 256 + n];
    } else {
      int e = idx - 8192 - 65536 - 65536, n = e / 288, k = e % 288;
      Wt4[n * 288 + k] = (_Float16)((k < 259) ? W4[k * 128 + n] : 0.f);
    }
  }
}

// ---------------- kernel 2: fused WMMA MLP over 32-point tiles ----------------
__global__ __launch_bounds__(256) void nerf_mlp_kernel(
    const float* __restrict__ rays_o, const float* __restrict__ rays_d,
    const float* __restrict__ t_rand,
    const float* __restrict__ b1, const float* __restrict__ b2, const float* __restrict__ b3,
    const float* __restrict__ Wd, const float* __restrict__ bd,
    const float* __restrict__ b4, const float* __restrict__ W5, const float* __restrict__ b5,
    const _Float16* __restrict__ Wt1, const _Float16* __restrict__ Wt2,
    const _Float16* __restrict__ Wt3, const _Float16* __restrict__ Wt4,
    float* __restrict__ dens_out, float* __restrict__ rgb_out)
{
  // LDS: strides padded by 8 halves (16B; hB by 40) to spread banks. Total ~47 KB.
  __shared__ __align__(16) _Float16 X [32 * 40];   // layer-1 input; later viewdirs tail
  __shared__ __align__(16) _Float16 hA[32 * 264];  // 256-wide activations (ping)
  __shared__ __align__(16) _Float16 hB[32 * 296];  // 256-wide activations (pong)
  __shared__ __align__(16) _Float16 h2[32 * 136];  // 128-wide color-head activations
  __shared__ float vd[32 * 3];

  const int tid   = threadIdx.x;
  const int wave  = tid >> 5;
  const int lane  = tid & 31;
  const int pbase = blockIdx.x * 32;

  for (int i = tid; i < 32 * 40; i += 256) X[i] = (_Float16)0.f;
  __syncthreads();

  if (tid < 32) {
    const int p = pbase + tid, r = p >> 6, s = p & (NS - 1);
    const float ox = rays_o[r * 3 + 0], oy = rays_o[r * 3 + 1], oz = rays_o[r * 3 + 2];
    const float dx = rays_d[r * 3 + 0], dy = rays_d[r * 3 + 1], dz = rays_d[r * 3 + 2];
    const float dstep = (FARV - NEARV) / (float)(NS - 1);
    const float zs    = NEARV + dstep * (float)s;
    const float lo    = (s == 0)      ? zs : zs - 0.5f * dstep;
    const float up    = (s == NS - 1) ? zs : zs + 0.5f * dstep;
    const float z     = lo + (up - lo) * t_rand[r * NS + s];
    X[tid * 40 + 0] = (_Float16)(ox + dx * z);
    X[tid * 40 + 1] = (_Float16)(oy + dy * z);
    X[tid * 40 + 2] = (_Float16)(oz + dz * z);
    const float nrm = sqrtf(dx * dx + dy * dy + dz * dz) + 1e-8f;
    vd[tid * 3 + 0] = dx / nrm; vd[tid * 3 + 1] = dy / nrm; vd[tid * 3 + 2] = dz / nrm;
  }
  __syncthreads();

  // layer 1: [32x32] @ [32x256]
  gemm_layer<2, 1>(X, 40, nullptr, 0, Wt1, 32, b1, hA, 264, wave, lane);
  __syncthreads();

  // repurpose X as the K=256..287 tail (viewdirs + zero pad) for layer 4
  if (tid < 32) {
    X[tid * 40 + 0] = (_Float16)vd[tid * 3 + 0];
    X[tid * 40 + 1] = (_Float16)vd[tid * 3 + 1];
    X[tid * 40 + 2] = (_Float16)vd[tid * 3 + 2];
  }

  // layer 2: [32x256] @ [256x256]
  gemm_layer<2, 8>(hA, 264, nullptr, 0, Wt2, 256, b2, hB, 296, wave, lane);
  __syncthreads();
  // layer 3: [32x256] @ [256x256]
  gemm_layer<2, 8>(hB, 296, nullptr, 0, Wt3, 256, b3, hA, 264, wave, lane);
  __syncthreads();

  // density head: [32x256] @ [256x1]; 8 lanes per point, shfl-xor tree reduce
  {
    const int i   = tid >> 3;   // point 0..31
    const int seg = tid & 7;    // 32-element K segment
    float part = 0.f;
#pragma unroll 4
    for (int k = seg * 32; k < seg * 32 + 32; ++k)
      part += (float)hA[i * 264 + k] * Wd[k];
    part += __shfl_xor(part, 1, 32);
    part += __shfl_xor(part, 2, 32);
    part += __shfl_xor(part, 4, 32);
    if (seg == 0) dens_out[pbase + i] = part + bd[0];
  }

  // layer 4: [32x288] @ [288x128], last K-tile from the viewdirs tail buffer
  gemm_layer<1, 9>(hA, 264, X, 40, Wt4, 288, b4, h2, 136, wave, lane);
  __syncthreads();

  // color head: [32x128] @ [128x3] + sigmoid; 2 lanes per (point, channel)
  if (tid < 192) {
    const int unit = tid >> 1;          // 0..95 = point*3 + channel
    const int seg  = tid & 1;           // 64-element K segment
    const int i = unit / 3, ch = unit % 3;
    float part = 0.f;
#pragma unroll 4
    for (int k = seg * 64; k < seg * 64 + 64; ++k)
      part += (float)h2[i * 136 + k] * W5[k * 3 + ch];
    part += __shfl_xor(part, 1, 32);
    if (seg == 0) {
      const float acc = part + b5[ch];
      rgb_out[(pbase + i) * 3 + ch] = 1.f / (1.f + expf(-acc));
    }
  }
}

// ---------------- kernel 3: per-ray alpha compositing ----------------
__global__ void composite_kernel(const float* __restrict__ t_rand,
                                 const float* __restrict__ dens,
                                 const float* __restrict__ rgb,
                                 float* __restrict__ out) {
  const int r = blockIdx.x * blockDim.x + threadIdx.x;
  if (r >= NRAYS) return;
  const float dstep = (FARV - NEARV) / (float)(NS - 1);

  auto zval = [&](int s) -> float {
    const float zs = NEARV + dstep * (float)s;
    const float lo = (s == 0)      ? zs : zs - 0.5f * dstep;
    const float up = (s == NS - 1) ? zs : zs + 0.5f * dstep;
    return lo + (up - lo) * t_rand[r * NS + s];
  };

  float T = 1.f, cr = 0.f, cg = 0.f, cb = 0.f, depth = 0.f, accw = 0.f;
  float zcur = zval(0);
  for (int s = 0; s < NS; ++s) {
    const float znext = (s < NS - 1) ? zval(s + 1) : 0.f;
    const float dist  = (s < NS - 1) ? (znext - zcur) : 1e10f;
    float d = dens[r * NS + s];
    d = d > 0.f ? d : 0.f;
    const float alpha = 1.f - expf(-d * dist);
    const float w = alpha * T;
    cr += w * rgb[(r * NS + s) * 3 + 0];
    cg += w * rgb[(r * NS + s) * 3 + 1];
    cb += w * rgb[(r * NS + s) * 3 + 2];
    depth += w * zcur;
    accw  += w;
    T *= (1.f - alpha + 1e-10f);
    zcur = znext;
  }
  const float bg = 1.f - accw;
  out[r * 3 + 0] = cr + bg;
  out[r * 3 + 1] = cg + bg;
  out[r * 3 + 2] = cb + bg;
  out[NRAYS * 3 + r] = depth;   // depth_map
  out[NRAYS * 4 + r] = accw;    // acc_map
}

extern "C" void kernel_launch(void* const* d_in, const int* in_sizes, int n_in,
                              void* d_out, int out_size, void* d_ws, size_t ws_size,
                              hipStream_t stream) {
  (void)in_sizes; (void)n_in; (void)out_size; (void)ws_size;
  const float* rays_o = (const float*)d_in[0];
  const float* rays_d = (const float*)d_in[1];
  const float* t_rand = (const float*)d_in[2];
  const float* W1 = (const float*)d_in[3];
  const float* b1 = (const float*)d_in[4];
  const float* W2 = (const float*)d_in[5];
  const float* b2 = (const float*)d_in[6];
  const float* W3 = (const float*)d_in[7];
  const float* b3 = (const float*)d_in[8];
  const float* Wd = (const float*)d_in[9];
  const float* bd = (const float*)d_in[10];
  const float* W4 = (const float*)d_in[11];
  const float* b4 = (const float*)d_in[12];
  const float* W5 = (const float*)d_in[13];
  const float* b5 = (const float*)d_in[14];

  char* ws = (char*)d_ws;
  _Float16* Wt1 = (_Float16*)(ws + WT1_OFF);
  _Float16* Wt2 = (_Float16*)(ws + WT2_OFF);
  _Float16* Wt3 = (_Float16*)(ws + WT3_OFF);
  _Float16* Wt4 = (_Float16*)(ws + WT4_OFF);
  float* dens = (float*)(ws + DEN_OFF);
  float* rgb  = (float*)(ws + RGB_OFF);

  prep_weights_kernel<<<688, 256, 0, stream>>>(W1, W2, W3, W4, Wt1, Wt2, Wt3, Wt4);

  nerf_mlp_kernel<<<NPTS / 32, 256, 0, stream>>>(
      rays_o, rays_d, t_rand, b1, b2, b3, Wd, bd, b4, W5, b5,
      Wt1, Wt2, Wt3, Wt4, dens, rgb);

  composite_kernel<<<(NRAYS + 255) / 256, 256, 0, stream>>>(t_rand, dens, rgb, (float*)d_out);
}
